// XTModel_46145128628499
// MI455X (gfx1250) — matrix operand order repaired and checked by reading, not compiled, not run on hardware
//
#include <hip/hip_runtime.h>
#include <math.h>

// Problem constants (match reference)
#define B_T   4096
#define DDIM  2048
#define VOC   50257
#define CAPV  30.0f
#define NTILE 256                  // vocab cols per workgroup
#define MTILE 128                  // tokens per workgroup
#define KSTEP 32                   // WMMA K
#define VT    ((VOC + NTILE - 1) / NTILE)   // 197 vocab tiles

typedef __attribute__((ext_vector_type(16))) _Float16 v16h;
typedef __attribute__((ext_vector_type(8)))  float    v8f;
typedef __attribute__((ext_vector_type(2)))  _Float16 h2;

// padded LDS row strides in halfs: 40 halfs = 80 B (16B-aligned rows)
#define ASTRIDE 40
#define BSTRIDE 40

union Frag { uint4 q[2]; v16h v; };

static __device__ __forceinline__ unsigned pk2(float a, float b) {
    h2 h = { (_Float16)a, (_Float16)b };         // -> v_cvt_pk_f16_f32
    return __builtin_bit_cast(unsigned, h);
}

// low 32 bits of a generic pointer to LDS == LDS byte offset (ISA aperture rule)
static __device__ __forceinline__ unsigned lds_addr(const void* p) {
    return (unsigned)(unsigned long long)p;
}

// async global -> LDS, 16 bytes per lane; IOFFSET applies to both sides
#define ASYNC_B128(ldsb, gaddr, OFF)                                          \
    asm volatile("global_load_async_to_lds_b128 %0, %1, off offset:" #OFF     \
                 :: "v"(ldsb), "v"(gaddr) : "memory")

static __device__ __forceinline__ void wait_async0() {
#if __has_builtin(__builtin_amdgcn_s_wait_asynccnt)
    __builtin_amdgcn_s_wait_asynccnt(0);
#else
    asm volatile("s_wait_asynccnt 0x0" ::: "memory");
#endif
}

// ---------------- phase 0a: hidden fp32 -> f16 (same [t][k] layout) -------
__global__ __launch_bounds__(256)
void cvt_hidden(const float* __restrict__ in, _Float16* __restrict__ out)
{
    const size_t i = ((size_t)blockIdx.x * 256 + threadIdx.x) * 4;
    const float4 f = *(const float4*)(in + i);
    uint2 p; p.x = pk2(f.x, f.y); p.y = pk2(f.z, f.w);
    *(uint2*)(out + i) = p;
}

// ---------------- phase 0b: lm_head [D][V] fp32 -> Bt [V][D] f16 ----------
__global__ __launch_bounds__(256)
void transpose_w(const float* __restrict__ lm, _Float16* __restrict__ bt)
{
    __shared__ float tile[32][33];
    const int n0 = blockIdx.x * 32;
    const int k0 = blockIdx.y * 32;
    const int tn = threadIdx.x & 31;
    const int tk = threadIdx.x >> 5;          // 0..7
    const int n  = n0 + tn;
    #pragma unroll
    for (int i = 0; i < 4; ++i) {
        const int k = tk + i * 8;
        tile[k][tn] = (n < VOC) ? lm[(size_t)(k0 + k) * VOC + n] : 0.f;
    }
    __syncthreads();
    const int on = threadIdx.x >> 4;          // 0..15
    const int kp = (threadIdx.x & 15) * 2;
    #pragma unroll
    for (int i = 0; i < 2; ++i) {
        const int nn = n0 + on + i * 16;
        if (nn < VOC) {
            const unsigned p = pk2(tile[kp][on + i * 16], tile[kp + 1][on + i * 16]);
            *(unsigned*)(bt + (size_t)nn * DDIM + k0 + kp) = p;
        }
    }
}

// ---------------- phase 1: WMMA GEMM + softcap + partial softmax ----------
__global__ __launch_bounds__(256)
void ce_gemm(const _Float16* __restrict__ hidf, const _Float16* __restrict__ btf,
             const int* __restrict__ targets,
             float* __restrict__ ms,   // [VT][B_T][2] (max, sumexp)
             float* __restrict__ tl)   // [B_T] softcapped target logit
{
    __shared__ __align__(16) _Float16 Ab[2][MTILE * ASTRIDE];  // 2 x 10240 B
    __shared__ __align__(16) _Float16 Bb[2][NTILE * BSTRIDE];  // 2 x 20480 B
    __shared__ float2 red[MTILE][4];                           // 4096 B

    const int tid    = threadIdx.x;
    const int m_base = blockIdx.x * MTILE;
    const int n_base = blockIdx.y * NTILE;

    const int wv   = tid >> 5;           // 8 waves in 2(m) x 4(n) grid
    const int lane = tid & 31;
    const int lmod = lane & 15;
    const int lhi  = lane >> 4;
    const int wm   = wv >> 2;            // 0..1 -> m offset wm*64
    const int wn   = wv & 3;             // 0..3 -> n offset wn*64

    // ---- staging assignments (per-thread, loop-invariant) ----
    // A: thread -> row m = tid/2, 32B half-slice (tid&1)
    const int am   = tid >> 1;
    const int aseg = tid & 1;
    const unsigned ldsA[2] = {
        lds_addr(&Ab[0][0]) + (unsigned)(am * 80 + aseg * 32),
        lds_addr(&Ab[1][0]) + (unsigned)(am * 80 + aseg * 32)
    };
    const unsigned long long gA =
        (unsigned long long)(const void*)(hidf + (size_t)(m_base + am) * DDIM + aseg * 16);
    // B: thread -> row n = tid (64B slice)
    const int n_g   = n_base + tid;
    const int n_src = (n_g < VOC) ? n_g : 0;   // clamp; masked in epilogue
    const unsigned ldsB[2] = {
        lds_addr(&Bb[0][0]) + (unsigned)(tid * 80),
        lds_addr(&Bb[1][0]) + (unsigned)(tid * 80)
    };
    const unsigned long long gB =
        (unsigned long long)(const void*)(btf + (size_t)n_src * DDIM);

    v8f acc[4][4];
    #pragma unroll
    for (int i = 0; i < 4; ++i)
        #pragma unroll
        for (int j = 0; j < 4; ++j)
            acc[i][j] = (v8f){0.f,0.f,0.f,0.f,0.f,0.f,0.f,0.f};

    auto stage = [&](int k0, int buf) {
        const unsigned la = ldsA[buf];
        const unsigned lb = ldsB[buf];
        const unsigned long long sa = gA + (unsigned long long)k0 * 2;
        const unsigned long long sb = gB + (unsigned long long)k0 * 2;
        ASYNC_B128(la, sa, 0);
        ASYNC_B128(la, sa, 16);
        ASYNC_B128(lb, sb, 0);
        ASYNC_B128(lb, sb, 16);
        ASYNC_B128(lb, sb, 32);
        ASYNC_B128(lb, sb, 48);
    };

    stage(0, 0);
    wait_async0();
    __syncthreads();

    int cur = 0;
    #pragma unroll 1
    for (int k0 = 0; k0 < DDIM; k0 += KSTEP) {
        if (k0 + KSTEP < DDIM) stage(k0 + KSTEP, cur ^ 1);   // prefetch next

        // B fragments: lane holds col n = lmod(+tile), K = lhi*16 + e
        Frag fb[4];
        #pragma unroll
        for (int nt = 0; nt < 4; ++nt) {
            const _Float16* br = &Bb[cur][(wn * 64 + nt * 16 + lmod) * BSTRIDE + lhi * 16];
            fb[nt].q[0] = ((const uint4*)br)[0];
            fb[nt].q[1] = ((const uint4*)br)[1];
        }
        // A fragments: lane holds row m = lmod(+tile), K = lhi*16 + e
        #pragma unroll
        for (int mt = 0; mt < 4; ++mt) {
            const _Float16* ar = &Ab[cur][(wm * 64 + mt * 16 + lmod) * ASTRIDE + lhi * 16];
            Frag fa;
            fa.q[0] = ((const uint4*)ar)[0];
            fa.q[1] = ((const uint4*)ar)[1];
            #pragma unroll
            for (int nt = 0; nt < 4; ++nt) {
                acc[mt][nt] = __builtin_amdgcn_wmma_f32_16x16x32_f16(
                    false, fa.v, false, fb[nt].v, (short)0, acc[mt][nt],
                    false, false);
            }
        }
        wait_async0();
        __syncthreads();
        cur ^= 1;
    }

    // ---- epilogue: softcap + per-row online softmax, in registers ----
    #pragma unroll
    for (int mt = 0; mt < 4; ++mt) {
        #pragma unroll
        for (int r = 0; r < 8; ++r) {
            const int m     = wm * 64 + mt * 16 + lhi * 8 + r;
            const int token = m_base + m;
            const int tgt   = targets[token];
            float M = -INFINITY;
            float vv[4];
            #pragma unroll
            for (int nt = 0; nt < 4; ++nt) {
                const int ng = n_base + wn * 64 + nt * 16 + lmod;
                float v = acc[mt][nt][r];
                v = CAPV * tanhf(v * (1.0f / CAPV));
                if (ng == tgt) tl[token] = v;            // unique writer
                v = (ng < VOC) ? v : -INFINITY;
                vv[nt] = v;
                M = fmaxf(M, v);
            }
            float S = 0.f;
            #pragma unroll
            for (int nt = 0; nt < 4; ++nt)
                S += (vv[nt] > -INFINITY) ? expf(vv[nt] - M) : 0.f;
            // reduce across the 16-lane half (xor<16 stays in half)
            #pragma unroll
            for (int off = 1; off < 16; off <<= 1) {
                const float Mo = __shfl_xor(M, off, 32);
                const float So = __shfl_xor(S, off, 32);
                const float nm = fmaxf(M, Mo);
                const float a = (S  > 0.f) ? S  * expf(M  - nm) : 0.f;
                const float b = (So > 0.f) ? So * expf(Mo - nm) : 0.f;
                M = nm; S = a + b;
            }
            if (lmod == 0) red[m][wn] = make_float2(M, S);
        }
    }
    __syncthreads();

    if (tid < MTILE) {
        const int token = m_base + tid;
        float M = -INFINITY, S = 0.f;
        #pragma unroll
        for (int w = 0; w < 4; ++w) {
            const float2 p = red[tid][w];
            const float nm = fmaxf(M, p.x);
            const float a = (S   > 0.f) ? S   * expf(M   - nm) : 0.f;
            const float b = (p.y > 0.f) ? p.y * expf(p.x - nm) : 0.f;
            M = nm; S = a + b;
        }
        const size_t idx = (size_t)blockIdx.y * B_T + token;
        ms[idx * 2]     = M;
        ms[idx * 2 + 1] = S;
    }
}

// ---------------- phase 2: merge partials -> per-token NLL ----------------
__global__ __launch_bounds__(256)
void ce_merge(const float* __restrict__ ms, const float* __restrict__ tl,
              float* __restrict__ loss)
{
    const int t = blockIdx.x * blockDim.x + threadIdx.x;
    float M = -INFINITY, S = 0.f;
    for (int vt = 0; vt < VT; ++vt) {
        const size_t idx = (size_t)vt * B_T + t;
        const float m = ms[idx * 2];
        const float s = ms[idx * 2 + 1];
        const float nm = fmaxf(M, m);
        const float a = (S > 0.f) ? S * expf(M - nm) : 0.f;
        const float b = (s > 0.f) ? s * expf(m - nm) : 0.f;
        M = nm; S = a + b;
    }
    loss[t] = (M + logf(S)) - tl[t];
}

// ---------------- phase 3: mean --------------------------------------------
__global__ __launch_bounds__(256)
void ce_mean(const float* __restrict__ loss, float* __restrict__ out)
{
    __shared__ float red[256];
    float a = 0.f;
    for (int i = threadIdx.x; i < B_T; i += 256) a += loss[i];
    red[threadIdx.x] = a;
    __syncthreads();
    for (int s = 128; s > 0; s >>= 1) {
        if (threadIdx.x < s) red[threadIdx.x] += red[threadIdx.x + s];
        __syncthreads();
    }
    if (threadIdx.x == 0) out[0] = red[0] * (1.0f / (float)B_T);
}

extern "C" void kernel_launch(void* const* d_in, const int* in_sizes, int n_in,
                              void* d_out, int out_size, void* d_ws, size_t ws_size,
                              hipStream_t stream)
{
    const float* hidden  = (const float*)d_in[0];
    const float* lm_head = (const float*)d_in[1];
    const int*   targets = (const int*)d_in[2];
    float* out = (float*)d_out;

    // workspace layout
    _Float16* btf  = (_Float16*)d_ws;                    // [VOC][DDIM] f16
    _Float16* hidf = btf + (size_t)VOC * DDIM;           // [B_T][DDIM] f16
    float*    ms   = (float*)(hidf + (size_t)B_T * DDIM);// [VT][B_T][2]
    float*    tl   = ms + (size_t)VT * B_T * 2;          // [B_T]
    float*    loss = tl + B_T;                           // [B_T]

    cvt_hidden<<<(B_T * DDIM / 4) / 256, 256, 0, stream>>>(hidden, hidf);
    transpose_w<<<dim3((VOC + 31) / 32, DDIM / 32), 256, 0, stream>>>(lm_head, btf);
    ce_gemm<<<dim3(B_T / MTILE, VT), 256, 0, stream>>>(hidf, btf, targets, ms, tl);
    ce_merge<<<B_T / 256, 256, 0, stream>>>(ms, tl, loss);
    ce_mean<<<1, 256, 0, stream>>>(loss, out);
}
